// GNOBlock_89730456748241
// MI455X (gfx1250) — matrix operand order repaired and checked by reading, not compiled, not run on hardware
//
#include <hip/hip_runtime.h>
#include <hip/hip_bf16.h>
#include <math.h>

typedef __attribute__((ext_vector_type(16))) _Float16 v16h;
typedef __attribute__((ext_vector_type(8)))  _Float16 v8h;
typedef __attribute__((ext_vector_type(4)))  _Float16 h4;
typedef __attribute__((ext_vector_type(8)))  float    v8f;

constexpr int N    = 10000;
constexpr int E    = 320000;
constexpr int B    = 2;
constexpr int CIN  = 64;
constexpr int COUT = 128;
constexpr int HID  = 128;
constexpr int KINP = 96;     // padded kin: [pos_d(3) pos_s(3) 0 0 x(64) 0*24]
constexpr int SMS  = 136;    // LDS row stride (halfs): 272B = 68 dwords -> conflict-free
constexpr int XSS  = 72;     // finalize LDS row stride (halfs): 144B = 36 dwords
constexpr float EPS = 1e-12f;

// ---------------------------------------------------------------------------
// gfx1250 async global->LDS copy (ASYNCcnt path), via inline asm.
// ---------------------------------------------------------------------------
__device__ __forceinline__ uint32_t lds_offset(const void* p) {
    // generic -> LDS addrspace cast; AS(3) pointers are 32-bit LDS byte offsets
    return (uint32_t)(uintptr_t)(__attribute__((address_space(3))) const void*)p;
}
__device__ __forceinline__ void async_load_b128(uint32_t lds_dst, const void* gaddr) {
    asm volatile("global_load_async_to_lds_b128 %0, %1, off"
                 :: "v"(lds_dst), "v"((uint64_t)(uintptr_t)gaddr)
                 : "memory");
}
__device__ __forceinline__ void wait_async0() {
    asm volatile("s_wait_asynccnt 0" ::: "memory");
}

// ---------------------------------------------------------------------------
// Branch-free exact-GELU: erf via Abramowitz-Stegun 7.1.26 (max err 1.5e-7).
// ---------------------------------------------------------------------------
__device__ __forceinline__ float fast_gelu(float x) {
    const float ax = fabsf(x) * 0.70710678118654752f;     // |x|/sqrt(2)
    const float t  = __builtin_amdgcn_rcpf(fmaf(0.3275911f, ax, 1.0f));
    float p = fmaf(1.061405429f, t, -1.453152027f);
    p = fmaf(p, t,  1.421413741f);
    p = fmaf(p, t, -0.284496736f);
    p = fmaf(p, t,  0.254829592f);
    p = p * t;
    const float e = __builtin_amdgcn_exp2f(-ax * ax * 1.4426950408889634f);
    const float erf_abs = fmaf(-p, e, 1.0f);
    const float erf_s   = copysignf(erf_abs, x);
    return 0.5f * x * (1.0f + erf_s);
}

// ---------------------------------------------------------------------------
// One MLP layer of WMMAs.
// ---------------------------------------------------------------------------
__device__ __forceinline__ void wmma_layer(const _Float16* __restrict__ abuf, int astride,
                                           const _Float16* __restrict__ wt, int wstride,
                                           int ksteps, v8f acc[8], int lane) {
    const int mr   = lane & 15;
    const int aoff = (lane < 16) ? 0 : 16;
    const int koff = (lane < 16) ? 0 : 16;
    for (int ks = 0; ks < ksteps; ++ks) {
        const int K = ks * 32;
        const _Float16* pa = abuf + mr * astride + K + aoff;
        v8h lo = *(const v8h*)pa;
        v8h hi = *(const v8h*)(pa + 8);
        v16h a = __builtin_shufflevector(lo, hi, 0,1,2,3,4,5,6,7,8,9,10,11,12,13,14,15);
#pragma unroll
        for (int ct = 0; ct < 8; ++ct) {
            const _Float16* pb = wt + (size_t)(ct * 16 + mr) * wstride + K + koff;
            v16h bf = *(const v16h*)pb;
            acc[ct] = __builtin_amdgcn_wmma_f32_16x16x32_f16(
                false, a, false, bf, (short)0, acc[ct], false, false);
        }
    }
}

// ---------------------------------------------------------------------------
// Kernel 0: zero agg + den
// ---------------------------------------------------------------------------
__global__ void zero_kernel(float* __restrict__ p, int count) {
    int i = blockIdx.x * blockDim.x + threadIdx.x;
    if (i < count) p[i] = 0.0f;
}

// ---------------------------------------------------------------------------
// Kernel 1a: convert + transpose weights to f16 column-major Wt[n][k].
// ---------------------------------------------------------------------------
__global__ void prep_weights(const float* __restrict__ W1, const float* __restrict__ W2,
                             const float* __restrict__ W3, const float* __restrict__ Ws,
                             _Float16* __restrict__ W1t, _Float16* __restrict__ W2t,
                             _Float16* __restrict__ W3t, _Float16* __restrict__ Wst) {
    int id = blockIdx.x * blockDim.x + threadIdx.x;
    if (id < HID * KINP) {                       // W1t: [128][96]
        int n = id / KINP, k = id % KINP;
        float v = 0.0f;
        if (k < 6)                 v = W1[k * HID + n];          // pos_dst, pos_src
        else if (k >= 8 && k < 72) v = W1[(k - 2) * HID + n];    // x features
        W1t[id] = (_Float16)v;
        return;
    }
    id -= HID * KINP;
    if (id < HID * HID) {                        // W2t: [128][128]
        int n = id / HID, k = id % HID;
        W2t[id] = (_Float16)W2[k * HID + n];
        return;
    }
    id -= HID * HID;
    if (id < COUT * HID) {                       // W3t: [128][128]
        int n = id / HID, k = id % HID;
        W3t[id] = (_Float16)W3[k * COUT + n];
        return;
    }
    id -= COUT * HID;
    if (id < COUT * CIN) {                       // Wst: [128][64]
        int n = id / CIN, k = id % CIN;
        Wst[id] = (_Float16)Ws[k * COUT + n];
        return;
    }
}

// ---------------------------------------------------------------------------
// Kernel 1b: x (f32) -> xh (f16), row layout unchanged
// ---------------------------------------------------------------------------
__global__ void prep_x(const float* __restrict__ x, _Float16* __restrict__ xh, int count) {
    int i = blockIdx.x * blockDim.x + threadIdx.x;
    if (i < count) xh[i] = (_Float16)x[i];
}

// ---------------------------------------------------------------------------
// Kernel 2: denominator segment sum
// ---------------------------------------------------------------------------
__global__ void den_kernel(const int* __restrict__ dst, const float* __restrict__ ew,
                           float* __restrict__ den) {
    int e = blockIdx.x * blockDim.x + threadIdx.x;
    if (e < E) unsafeAtomicAdd(&den[dst[e]], ew[e]);
}

// ---------------------------------------------------------------------------
// Kernel 3: edge MLP — one 16-edge tile per wave
// ---------------------------------------------------------------------------
#define WPB 4  // waves per block
__global__ __launch_bounds__(WPB * 32) void edge_mlp_kernel(
    const _Float16* __restrict__ xh, const float* __restrict__ pos,
    const int* __restrict__ src, const int* __restrict__ dst,
    const float* __restrict__ ew,
    const float* __restrict__ b1, const float* __restrict__ b2, const float* __restrict__ b3,
    const _Float16* __restrict__ W1t, const _Float16* __restrict__ W2t,
    const _Float16* __restrict__ W3t,
    float* __restrict__ agg) {

    __shared__ _Float16 sm[WPB][2][16 * SMS];

    const int wave = threadIdx.x >> 5;
    const int lane = threadIdx.x & 31;
    const int tile = blockIdx.x * WPB + wave;    // E/16 = 20000 exact, no tail
    const int b    = blockIdx.y;
    const int tb   = tile * 16;

    // ---- stage kin tile: x rows via async global->LDS, pos/pad via DS ----
    {
        _Float16* kin = sm[wave][0];
        const int m    = lane & 15;              // edge within tile
        const int half = lane >> 4;              // which 32-half chunk of x row
        const int e    = tb + m;
        const int s    = src[e];

        // x features (already f16) -> kin[m][8 + half*32 .. +31]
        const _Float16* xrow = xh + ((size_t)b * N + s) * CIN + half * 32;
        const uint32_t  ldst = lds_offset(&kin[m * SMS + 8 + half * 32]);
#pragma unroll
        for (int q = 0; q < 4; ++q)
            async_load_b128(ldst + q * 16, xrow + q * 8);

        if (half == 0) {
            // pos_dst(3), pos_src(3), two zeros -> kin[m][0..7]
            const int d = dst[e];
            float pd0 = pos[d * 3 + 0], pd1 = pos[d * 3 + 1], pd2 = pos[d * 3 + 2];
            float ps0 = pos[s * 3 + 0], ps1 = pos[s * 3 + 1], ps2 = pos[s * 3 + 2];
            h4 a = { (_Float16)pd0, (_Float16)pd1, (_Float16)pd2, (_Float16)ps0 };
            h4 c = { (_Float16)ps1, (_Float16)ps2, (_Float16)0.0f, (_Float16)0.0f };
            *(h4*)&kin[m * SMS + 0] = a;
            *(h4*)&kin[m * SMS + 4] = c;
        } else {
            // zero pad kin[m][72..95]
            h4 z = { (_Float16)0.0f, (_Float16)0.0f, (_Float16)0.0f, (_Float16)0.0f };
#pragma unroll
            for (int q = 0; q < 6; ++q)
                *(h4*)&kin[m * SMS + 72 + q * 4] = z;
        }
    }
    wait_async0();
    __syncthreads();

    const int nl     = lane & 15;
    const int rowoff = (lane < 16) ? 0 : 8;

    v8f acc[8];
#pragma unroll
    for (int ct = 0; ct < 8; ++ct) acc[ct] = (v8f){0,0,0,0,0,0,0,0};

    // ---- layer 1: kin(16x96) @ W1 -> h1(16x128), gelu ----
    wmma_layer(sm[wave][0], SMS, W1t, KINP, 3, acc, lane);
#pragma unroll
    for (int ct = 0; ct < 8; ++ct) {
        const float bias = b1[ct * 16 + nl];
#pragma unroll
        for (int r = 0; r < 8; ++r) {
            float g = fast_gelu(acc[ct][r] + bias);
            sm[wave][1][(r + rowoff) * SMS + ct * 16 + nl] = (_Float16)g;
        }
        acc[ct] = (v8f){0,0,0,0,0,0,0,0};
    }
    __syncthreads();

    // ---- layer 2: h1 @ W2 -> h2, gelu ----
    wmma_layer(sm[wave][1], SMS, W2t, HID, 4, acc, lane);
#pragma unroll
    for (int ct = 0; ct < 8; ++ct) {
        const float bias = b2[ct * 16 + nl];
#pragma unroll
        for (int r = 0; r < 8; ++r) {
            float g = fast_gelu(acc[ct][r] + bias);
            sm[wave][0][(r + rowoff) * SMS + ct * 16 + nl] = (_Float16)g;
        }
        acc[ct] = (v8f){0,0,0,0,0,0,0,0};
    }
    __syncthreads();

    // ---- layer 3: h2 @ W3 -> msg ----
    wmma_layer(sm[wave][0], SMS, W3t, HID, 4, acc, lane);

    // ---- scale by edge weight and scatter-add into agg ----
    float  ww[8];
    float* base[8];
#pragma unroll
    for (int r = 0; r < 8; ++r) {
        const int e = tb + r + rowoff;
        ww[r]   = ew[e];
        base[r] = agg + ((size_t)b * N + dst[e]) * COUT + nl;
    }
#pragma unroll
    for (int ct = 0; ct < 8; ++ct) {
        const float bias = b3[ct * 16 + nl];
#pragma unroll
        for (int r = 0; r < 8; ++r) {
            const float val = (acc[ct][r] + bias) * ww[r];
            unsafeAtomicAdd(base[r] + ct * 16, val);   // immediate-offset atomics
        }
    }
}

// ---------------------------------------------------------------------------
// Kernel 4: finalize — out = agg/(den+eps) + x@Ws + bs (skip via WMMA).
// ---------------------------------------------------------------------------
__global__ __launch_bounds__(32) void finalize_kernel(
    const _Float16* __restrict__ xh, const float* __restrict__ bs,
    const _Float16* __restrict__ Wst,
    const float* __restrict__ agg, const float* __restrict__ den,
    float* __restrict__ out) {

    __shared__ _Float16 xs[16 * XSS];

    const int lane = threadIdx.x;
    const int tile = blockIdx.x;
    const int b    = blockIdx.y;

    // stage x tile (16 nodes x 64 halfs) via async copy
    {
        const int m    = lane & 15;
        const int half = lane >> 4;
        const int node = tile * 16 + m;
        const _Float16* xrow = xh + ((size_t)b * N + node) * CIN + half * 32;
        const uint32_t  ldst = lds_offset(&xs[m * XSS + half * 32]);
#pragma unroll
        for (int q = 0; q < 4; ++q)
            async_load_b128(ldst + q * 16, xrow + q * 8);
    }
    wait_async0();
    __syncthreads();

    v8f acc[8];
#pragma unroll
    for (int ct = 0; ct < 8; ++ct) acc[ct] = (v8f){0,0,0,0,0,0,0,0};

    wmma_layer(xs, XSS, Wst, CIN, 2, acc, lane);

    const int nl     = lane & 15;
    const int rowoff = (lane < 16) ? 0 : 8;

    float dinv[8];
#pragma unroll
    for (int r = 0; r < 8; ++r) {
        const int node = tile * 16 + r + rowoff;
        dinv[r] = 1.0f / (den[node] + EPS);
    }
#pragma unroll
    for (int ct = 0; ct < 8; ++ct) {
        const float bias = bs[ct * 16 + nl];
#pragma unroll
        for (int r = 0; r < 8; ++r) {
            const size_t idx = ((size_t)b * N + tile * 16 + r + rowoff) * COUT + ct * 16 + nl;
            out[idx] = acc[ct][r] + bias + agg[idx] * dinv[r];
        }
    }
}

// ---------------------------------------------------------------------------
// Launch
// ---------------------------------------------------------------------------
extern "C" void kernel_launch(void* const* d_in, const int* in_sizes, int n_in,
                              void* d_out, int out_size, void* d_ws, size_t ws_size,
                              hipStream_t stream) {
    const float* x   = (const float*)d_in[0];
    const float* pos = (const float*)d_in[1];
    const int*   ei  = (const int*)  d_in[2];
    const float* ew  = (const float*)d_in[3];
    const float* W1  = (const float*)d_in[4];
    const float* b1  = (const float*)d_in[5];
    const float* W2  = (const float*)d_in[6];
    const float* b2  = (const float*)d_in[7];
    const float* W3  = (const float*)d_in[8];
    const float* b3  = (const float*)d_in[9];
    const float* Ws  = (const float*)d_in[10];
    const float* bs  = (const float*)d_in[11];
    float* out = (float*)d_out;

    const int* src = ei;
    const int* dst = ei + E;

    // workspace layout (all 16B+ aligned by construction)
    float*     agg = (float*)d_ws;                       // B*N*COUT
    float*     den = agg + (size_t)B * N * COUT;         // N
    _Float16* W1t = (_Float16*)(den + N);                // 128*96
    _Float16* W2t = W1t + HID * KINP;                    // 128*128
    _Float16* W3t = W2t + HID * HID;                     // 128*128
    _Float16* Wst = W3t + COUT * HID;                    // 128*64
    _Float16* xh  = Wst + COUT * CIN;                    // B*N*CIN f16

    // 0) zero agg + den
    {
        const int count = B * N * COUT + N;
        zero_kernel<<<(count + 255) / 256, 256, 0, stream>>>(agg, count);
    }
    // 1a) weight prep
    {
        const int total = HID * KINP + HID * HID + COUT * HID + COUT * CIN;
        prep_weights<<<(total + 255) / 256, 256, 0, stream>>>(W1, W2, W3, Ws,
                                                              W1t, W2t, W3t, Wst);
    }
    // 1b) x -> f16
    {
        const int count = B * N * CIN;
        prep_x<<<(count + 255) / 256, 256, 0, stream>>>(x, xh, count);
    }
    // 2) denominator
    den_kernel<<<(E + 255) / 256, 256, 0, stream>>>(dst, ew, den);

    // 3) edge MLP: 20000 tiles, WPB per block, y = batch
    {
        dim3 grid((E / 16) / WPB, B, 1);
        edge_mlp_kernel<<<grid, WPB * 32, 0, stream>>>(xh, pos, src, dst, ew,
                                                       b1, b2, b3,
                                                       W1t, W2t, W3t, agg);
    }
    // 4) finalize: 625 node tiles, y = batch
    {
        dim3 grid(N / 16, B, 1);
        finalize_kernel<<<grid, 32, 0, stream>>>(xh, bs, Wst, agg, den, out);
    }
}
